// LSTMLoRAEncoder_85426899517665
// MI455X (gfx1250) — compile-verified, matrix operand order checked by reading, not run on hardware
//
#include <hip/hip_runtime.h>
#include <hip/hip_bf16.h>

typedef __bf16 bf16_t;
typedef __attribute__((ext_vector_type(16))) __bf16 v16bf;
typedef __attribute__((ext_vector_type(8)))  float  v8f;

#define LORA_SCALE 2.0f     // ALPHA/R = 16/8
#define T_STEPS 512
#define NBATCH  64
#define HID     1024
#define NGATES  4096

union FragU { v16bf v; uint4 q[2]; };

// Load one 16x32 bf16 WMMA operand fragment from a row-major [rows x ld] matrix.
// Per ISA 7.12.2 (16-bit A 16x32): lanes 0-15 hold row (row0+lane), K chunks {k0..k0+7, k0+16..k0+23};
// lanes 16-31 hold the same rows, K chunks {k0+8..15, k0+24..31}. B uses the same pattern on W (=B^T).
__device__ __forceinline__ v16bf load_frag(const bf16_t* base, int row0, int k0, int ld) {
    const int lane = threadIdx.x & 31;
    const int r  = row0 + (lane & 15);
    const int kk = k0 + ((lane >> 4) << 3);
    const bf16_t* p = base + (size_t)r * (size_t)ld + kk;
    FragU f;
    f.q[0] = *(const uint4*)(p);
    f.q[1] = *(const uint4*)(p + 16);
    return f.v;
}

__device__ __forceinline__ v8f zero8() {
    v8f z;
#pragma unroll
    for (int i = 0; i < 8; ++i) z[i] = 0.0f;
    return z;
}

__device__ __forceinline__ float fast_sigmoid(float x) {
    return 1.0f / (1.0f + __expf(-x));
}
__device__ __forceinline__ float fast_tanh(float x) {
    return 1.0f - 2.0f / (__expf(2.0f * x) + 1.0f);
}

// ---------------------------------------------------------------------------
// Fold LoRA into dense weights, convert to bf16:  We[n,k] = W[n,k] + s*B[n,:]@A[:,k]
// ---------------------------------------------------------------------------
__global__ void fold_weights_kernel(const float* __restrict__ W,
                                    const float* __restrict__ Bm,   // [N,8]
                                    const float* __restrict__ Am,   // [8,K]
                                    bf16_t* __restrict__ We,
                                    int N, int K) {
    int idx = blockIdx.x * 256 + threadIdx.x;
    if (idx >= N * K) return;
    int n = idx / K, k = idx - n * K;
    float acc = W[idx];
#pragma unroll
    for (int r = 0; r < 8; ++r)
        acc += LORA_SCALE * Bm[n * 8 + r] * Am[r * K + k];
    We[idx] = (bf16_t)acc;
}

// ---------------------------------------------------------------------------
// Embedding lookup -> bf16 sequence [T*B, 512] (row = t*64 + b)
// ---------------------------------------------------------------------------
__global__ void embed_kernel(const int* __restrict__ x,      // [B,T]
                             const float* __restrict__ emb,  // [VOCAB,512]
                             bf16_t* __restrict__ seq) {
    int row = blockIdx.x;            // t*64 + b
    int t = row >> 6, b = row & 63;
    int tok = x[b * T_STEPS + t];
    const float* src = emb + (size_t)tok * 512;
    bf16_t* dst = seq + (size_t)row * 512;
    for (int e = threadIdx.x; e < 512; e += 256)
        dst[e] = (bf16_t)src[e];
}

// ---------------------------------------------------------------------------
// C[M,4096] (f32) = A[M,K] (bf16, row-major) x W[4096,K]^T (bf16, row-major)
// Block = 256 threads = 8 waves; block tile 128x128; wave tile 64x32 (8 WMMA accums).
// ---------------------------------------------------------------------------
__global__ void wmma_gemm_bf16(const bf16_t* __restrict__ A,
                               const bf16_t* __restrict__ W,
                               float* __restrict__ C, int K) {
    const int wave = threadIdx.x >> 5;
    const int wm = wave & 1;       // 2 wave rows
    const int wn = wave >> 1;      // 4 wave cols
    const int m0 = blockIdx.x * 128 + wm * 64;
    const int n0 = blockIdx.y * 128 + wn * 32;

    v8f acc[4][2];
#pragma unroll
    for (int i = 0; i < 4; ++i)
#pragma unroll
        for (int j = 0; j < 2; ++j) acc[i][j] = zero8();

    for (int k0 = 0; k0 < K; k0 += 32) {
        v16bf a[4], b[2];
#pragma unroll
        for (int i = 0; i < 4; ++i) a[i] = load_frag(A, m0 + 16 * i, k0, K);
#pragma unroll
        for (int j = 0; j < 2; ++j) b[j] = load_frag(W, n0 + 16 * j, k0, K);
#pragma unroll
        for (int i = 0; i < 4; ++i)
#pragma unroll
            for (int j = 0; j < 2; ++j)
                acc[i][j] = __builtin_amdgcn_wmma_f32_16x16x32_bf16(
                    false, a[i], false, b[j], (short)0, acc[i][j], false, false);
    }

    const int lane = threadIdx.x & 31;
    const int mo = (lane >> 4) << 3;
    const int no = lane & 15;
#pragma unroll
    for (int i = 0; i < 4; ++i)
#pragma unroll
        for (int j = 0; j < 2; ++j) {
            float* cp = C + (size_t)(m0 + 16 * i + mo) * NGATES + (n0 + 16 * j + no);
#pragma unroll
            for (int g = 0; g < 8; ++g)
                cp[(size_t)g * NGATES] = acc[i][j][g];
        }
}

// ---------------------------------------------------------------------------
// Grid-wide barrier: monotonic counter, agent-scope release/acquire.
// ---------------------------------------------------------------------------
__device__ __forceinline__ void grid_sync(unsigned int* cnt, unsigned int nb,
                                          unsigned int phase) {
    __threadfence();
    __syncthreads();
    if (threadIdx.x == 0) {
        __hip_atomic_fetch_add(cnt, 1u, __ATOMIC_RELEASE, __HIP_MEMORY_SCOPE_AGENT);
        const unsigned int target = phase * nb;
        while (__hip_atomic_load(cnt, __ATOMIC_ACQUIRE, __HIP_MEMORY_SCOPE_AGENT) < target)
            __builtin_amdgcn_s_sleep(1);
    }
    __syncthreads();
    __threadfence();
}

__global__ void reset_sync_kernel(unsigned int* p) {
    if (threadIdx.x == 0) *p = 0u;
}

// ---------------------------------------------------------------------------
// Persistent LSTM scan. Grid = 64 blocks x 256 threads; block b owns h cols
// [16b, 16b+16) and its 64 gate rows of W_hh_eff, which stay resident in LDS
// (64 x 1024 bf16, padded) for all 512 steps. Per step:
//   - issue this step's xproj loads first (latency hidden under WMMA),
//   - 8 waves compute the 64x64 gate tile (h @ Whh_slice^T) with WMMA,
//   - stage it in LDS, fused elementwise LSTM cell with c in registers,
//   - prefetch next step's xproj tile, grid-wide barrier.
// ---------------------------------------------------------------------------
__global__ void lstm_scan_kernel(const bf16_t* __restrict__ Whh,   // [4096,1024] folded bf16
                                 const float*  __restrict__ xproj, // [T*64,4096] f32
                                 const float*  __restrict__ b_ih,
                                 const float*  __restrict__ b_hh,
                                 const int*    __restrict__ lengths,
                                 bf16_t* __restrict__ hbuf,        // [64,1024] bf16
                                 bf16_t* __restrict__ hseq,        // [T*64,1024] or null
                                 float*  __restrict__ out,         // [64,1024] or null
                                 unsigned int* __restrict__ sync) {
    extern __shared__ char smem[];
    const int SB = 1032;                       // bf16 row stride (+8 pad -> bank spread)
    const int SG = 65;                         // f32 gate-tile row stride
    bf16_t* Bl = (bf16_t*)smem;                // [64][1032] bf16 = 132096 B
    float*  gl = (float*)(smem + 64 * SB * sizeof(bf16_t)); // [64][65] f32 = 16640 B

    const int hc0 = blockIdx.x * 16;
    const int tid = threadIdx.x;
    const unsigned int nb = gridDim.x;

    // Load this block's 64 W_hh rows (i/f/g/o x 16 cols) into LDS, 16B chunks.
    for (int idx = tid; idx < 64 * 128; idx += 256) {
        int row = idx >> 7, chunk = idx & 127;
        int grow = (row >> 4) * HID + hc0 + (row & 15);
        *((uint4*)(Bl + row * SB) + chunk) =
            *((const uint4*)(Whh + (size_t)grow * HID) + chunk);
    }
    // Zero this block's slice of h.
    for (int idx = tid; idx < 64 * 16; idx += 256) {
        int m = idx >> 4, n = idx & 15;
        hbuf[(size_t)m * HID + hc0 + n] = (bf16_t)0.0f;
    }

    // Elementwise thread mapping: thread owns (m = tid>>2, 4 consecutive h cols).
    const int em  = tid >> 2;
    const int en0 = (tid & 3) * 4;
    float c_reg[4] = {0.f, 0.f, 0.f, 0.f};
    float bias[4][4];
#pragma unroll
    for (int g2 = 0; g2 < 4; ++g2)
#pragma unroll
        for (int j = 0; j < 4; ++j) {
            int col = g2 * HID + hc0 + en0 + j;
            bias[g2][j] = b_ih[col] + b_hh[col];
        }
    int lastt = lengths[em] - 1;
    if (lastt < 0) lastt = 0;

    // Per-thread base of this thread's 4-wide xproj segment (per gate: + g*HID).
    const float* xp_base = xproj + (size_t)em * NGATES + hc0 + en0;

    // WMMA wave mapping: wave -> (m-tile, n-tile-pair). n-tile == gate index.
    const int wave = tid >> 5, lane = tid & 31;
    const int mt = wave & 3, ntp = wave >> 2;

    grid_sync(sync, nb, 1);   // h zeros visible everywhere

    for (int t = 0; t < T_STEPS; ++t) {
        // --- Issue this step's xproj gate loads up-front; latency overlaps WMMA.
        const float* xp = xp_base + (size_t)t * NBATCH * NGATES;
        float4 xpv[4];
#pragma unroll
        for (int g2 = 0; g2 < 4; ++g2)
            xpv[g2] = *(const float4*)(xp + (size_t)g2 * HID);

        // --- WMMA: gate tile partials = h @ Whh_slice^T
        v8f acc0 = zero8(), acc1 = zero8();
#pragma unroll 4
        for (int ks = 0; ks < 32; ++ks) {
            const int k0 = ks * 32;
            v16bf a  = load_frag(hbuf, mt * 16, k0, HID);
            v16bf b0 = load_frag(Bl, (2 * ntp) * 16, k0, SB);
            v16bf b1 = load_frag(Bl, (2 * ntp + 1) * 16, k0, SB);
            acc0 = __builtin_amdgcn_wmma_f32_16x16x32_bf16(false, a, false, b0,
                                                           (short)0, acc0, false, false);
            acc1 = __builtin_amdgcn_wmma_f32_16x16x32_bf16(false, a, false, b1,
                                                           (short)0, acc1, false, false);
        }
        {   // stage gate tile in LDS: gl[m][gate*16 + nloc]
            const int mrow  = mt * 16 + ((lane >> 4) << 3);
            const int ncol0 = (2 * ntp) * 16 + (lane & 15);
#pragma unroll
            for (int g = 0; g < 8; ++g) {
                gl[(mrow + g) * SG + ncol0]      = acc0[g];
                gl[(mrow + g) * SG + ncol0 + 16] = acc1[g];
            }
        }
        // Prefetch next step's xproj tile while waves drain into the barrier.
        if (t + 1 < T_STEPS) {
            const float* xpn = xp + (size_t)NBATCH * NGATES;
#pragma unroll
            for (int g2 = 0; g2 < 4; ++g2)
                __builtin_prefetch(xpn + (size_t)g2 * HID, 0, 3);
        }
        __syncthreads();

        // --- Fused LSTM cell.
#pragma unroll
        for (int j = 0; j < 4; ++j) {
            const int nloc = en0 + j;
            float gi = gl[em * SG +  0 + nloc] + xpv[0][j] + bias[0][j];
            float gf = gl[em * SG + 16 + nloc] + xpv[1][j] + bias[1][j];
            float gg = gl[em * SG + 32 + nloc] + xpv[2][j] + bias[2][j];
            float go = gl[em * SG + 48 + nloc] + xpv[3][j] + bias[3][j];
            float iv = fast_sigmoid(gi);
            float fv = fast_sigmoid(gf);
            float gv = fast_tanh(gg);
            float ov = fast_sigmoid(go);
            float c  = fv * c_reg[j] + iv * gv;
            c_reg[j] = c;
            float h  = ov * fast_tanh(c);
            const int col = hc0 + nloc;
            hbuf[(size_t)em * HID + col] = (bf16_t)h;
            if (hseq) hseq[((size_t)t * NBATCH + em) * HID + col] = (bf16_t)h;
            if (out && t == lastt) out[(size_t)em * HID + col] = h;
        }
        grid_sync(sync, nb, (unsigned int)t + 2);
    }
}

// ---------------------------------------------------------------------------
extern "C" void kernel_launch(void* const* d_in, const int* in_sizes, int n_in,
                              void* d_out, int out_size, void* d_ws, size_t ws_size,
                              hipStream_t stream) {
    (void)in_sizes; (void)n_in; (void)out_size; (void)ws_size;

    const int*   x       = (const int*)d_in[0];
    const int*   lengths = (const int*)d_in[1];
    const float* emb     = (const float*)d_in[2];
    const float* W_ih0 = (const float*)d_in[3];
    const float* b_ih0 = (const float*)d_in[4];
    const float* A_ih0 = (const float*)d_in[5];
    const float* B_ih0 = (const float*)d_in[6];
    const float* W_hh0 = (const float*)d_in[7];
    const float* b_hh0 = (const float*)d_in[8];
    const float* A_hh0 = (const float*)d_in[9];
    const float* B_hh0 = (const float*)d_in[10];
    const float* W_ih1 = (const float*)d_in[11];
    const float* b_ih1 = (const float*)d_in[12];
    const float* A_ih1 = (const float*)d_in[13];
    const float* B_ih1 = (const float*)d_in[14];
    const float* W_hh1 = (const float*)d_in[15];
    const float* b_hh1 = (const float*)d_in[16];
    const float* A_hh1 = (const float*)d_in[17];
    const float* B_hh1 = (const float*)d_in[18];
    float* out = (float*)d_out;

    const size_t MROWS = (size_t)T_STEPS * NBATCH;   // 32768
    char* ws = (char*)d_ws;
    float*  xproj = (float*)ws;   ws += MROWS * NGATES * sizeof(float);   // 512 MB (reused)
    bf16_t* seq0  = (bf16_t*)ws;  ws += MROWS * 512 * sizeof(bf16_t);     // 32 MB
    bf16_t* hseq  = (bf16_t*)ws;  ws += MROWS * HID * sizeof(bf16_t);     // 64 MB
    bf16_t* Wih0e = (bf16_t*)ws;  ws += (size_t)NGATES * 512 * sizeof(bf16_t);
    bf16_t* Whh0e = (bf16_t*)ws;  ws += (size_t)NGATES * HID * sizeof(bf16_t);
    bf16_t* Wih1e = (bf16_t*)ws;  ws += (size_t)NGATES * HID * sizeof(bf16_t);
    bf16_t* Whh1e = (bf16_t*)ws;  ws += (size_t)NGATES * HID * sizeof(bf16_t);
    bf16_t* hbuf  = (bf16_t*)ws;  ws += (size_t)NBATCH * HID * sizeof(bf16_t);
    unsigned int* sync = (unsigned int*)ws;

    // 1) Fold LoRA into dense bf16 weights.
    fold_weights_kernel<<<(NGATES * 512 + 255) / 256, 256, 0, stream>>>(
        W_ih0, B_ih0, A_ih0, Wih0e, NGATES, 512);
    fold_weights_kernel<<<(NGATES * HID + 255) / 256, 256, 0, stream>>>(
        W_hh0, B_hh0, A_hh0, Whh0e, NGATES, HID);
    fold_weights_kernel<<<(NGATES * HID + 255) / 256, 256, 0, stream>>>(
        W_ih1, B_ih1, A_ih1, Wih1e, NGATES, HID);
    fold_weights_kernel<<<(NGATES * HID + 255) / 256, 256, 0, stream>>>(
        W_hh1, B_hh1, A_hh1, Whh1e, NGATES, HID);

    // 2) Embedding lookup -> bf16 [T*B, 512].
    embed_kernel<<<(int)MROWS, 256, 0, stream>>>(x, emb, seq0);

    const size_t smem = (size_t)64 * 1032 * sizeof(bf16_t) + (size_t)64 * 65 * sizeof(float);

    // 3) Layer 0: x_proj GEMM, then persistent recurrent scan.
    wmma_gemm_bf16<<<dim3(256, 32), 256, 0, stream>>>(seq0, Wih0e, xproj, 512);
    reset_sync_kernel<<<1, 1, 0, stream>>>(sync);
    lstm_scan_kernel<<<64, 256, smem, stream>>>(Whh0e, xproj, b_ih0, b_hh0, lengths,
                                                hbuf, hseq, nullptr, sync);

    // 4) Layer 1: x_proj GEMM from h-sequence, scan writes last_h to d_out.
    wmma_gemm_bf16<<<dim3(256, 32), 256, 0, stream>>>(hseq, Wih1e, xproj, 1024);
    reset_sync_kernel<<<1, 1, 0, stream>>>(sync);
    lstm_scan_kernel<<<64, 256, smem, stream>>>(Whh1e, xproj, b_ih1, b_hh1, lengths,
                                                hbuf, nullptr, out, sync);
}
